// TransformerBlock_6098853560447
// MI455X (gfx1250) — matrix operand order repaired and checked by reading, not compiled, not run on hardware
//
#include <hip/hip_runtime.h>

// ---------------------------------------------------------------------------
// CDNA5 (gfx1250) transformer block: bf16 WMMA GEMMs, fp32 accumulate.
// B=8, S=2048, E=EI=1024.
// GEMM: 8 waves/block, wave tile 64(M)x32(N) = 4x2 WMMA subtiles,
// block tile 128x128, K stepped by 32 -> 8 v_wmma per 12 b128 loads.
// ---------------------------------------------------------------------------

typedef __attribute__((ext_vector_type(16))) __bf16 v16bf;
typedef __attribute__((ext_vector_type(8)))  float  v8f;

#define E_DIM 1024
#define S_DIM 2048
#define B_DIM 8
#define M_TOT (B_DIM * S_DIM)   // 16384

__device__ __forceinline__ unsigned short f2bf(float f) {
  unsigned int u = __float_as_uint(f);
  u += 0x7FFFu + ((u >> 16) & 1u);           // round-to-nearest-even
  return (unsigned short)(u >> 16);
}
__device__ __forceinline__ float bf2f(unsigned short h) {
  return __uint_as_float(((unsigned int)h) << 16);
}

union FragBF {                 // 16 bf16 = 32 bytes = two b128 loads
  uint4 u[2];
  v16bf v;
};
union PackBF8 {                // 8 bf16 packed for one b128 store
  unsigned short s[8];
  uint4 q;
};

// ---------------------------------------------------------------------------
// Weight transpose + fp32->bf16 convert:  WT[c][r] = (bf16) W[r][c]
// ---------------------------------------------------------------------------
__global__ __launch_bounds__(256) void wt_convert_kernel(
    const float* __restrict__ W, unsigned short* __restrict__ WT, int R, int C) {
  __shared__ float tile[32][33];
  const int c0 = blockIdx.x * 32, r0 = blockIdx.y * 32;
  const int tx = threadIdx.x & 31, ty = threadIdx.x >> 5;
#pragma unroll
  for (int i = 0; i < 32; i += 8)
    tile[ty + i][tx] = W[(long)(r0 + ty + i) * C + (c0 + tx)];
  __syncthreads();
#pragma unroll
  for (int i = 0; i < 32; i += 8)
    WT[(long)(c0 + ty + i) * R + (r0 + tx)] = f2bf(tile[tx][ty + i]);
}

// ---------------------------------------------------------------------------
// LayerNorm over last dim (1024), one block (256 thr) per row, bf16 output.
// IN_BF16 = 0: fp32 input; 1: bf16 (ushort) input.
// ---------------------------------------------------------------------------
template <int IN_BF16>
__global__ __launch_bounds__(256) void layernorm_kernel(
    const void* __restrict__ xin, const float* __restrict__ g,
    const float* __restrict__ bta, unsigned short* __restrict__ out) {
  const int tid = threadIdx.x;
  const long base = (long)blockIdx.x * E_DIM;
  float v[4];
#pragma unroll
  for (int i = 0; i < 4; ++i) {
    const int idx = tid + i * 256;
    v[i] = IN_BF16 ? bf2f(((const unsigned short*)xin)[base + idx])
                   : ((const float*)xin)[base + idx];
  }
  __shared__ float redA[8], redB[8];
  float s = v[0] + v[1] + v[2] + v[3];
#pragma unroll
  for (int off = 16; off; off >>= 1) s += __shfl_xor(s, off, 32);
  if ((tid & 31) == 0) redA[tid >> 5] = s;
  __syncthreads();
  float tot = 0.f;
#pragma unroll
  for (int w = 0; w < 8; ++w) tot += redA[w];
  const float mu = tot * (1.0f / E_DIM);

  float var = 0.f;
#pragma unroll
  for (int i = 0; i < 4; ++i) { const float d = v[i] - mu; var += d * d; }
#pragma unroll
  for (int off = 16; off; off >>= 1) var += __shfl_xor(var, off, 32);
  if ((tid & 31) == 0) redB[tid >> 5] = var;
  __syncthreads();
  float vt = 0.f;
#pragma unroll
  for (int w = 0; w < 8; ++w) vt += redB[w];
  const float inv = rsqrtf(vt * (1.0f / E_DIM) + 1e-5f);
#pragma unroll
  for (int i = 0; i < 4; ++i) {
    const int idx = tid + i * 256;
    out[base + idx] = f2bf((v[i] - mu) * inv * g[idx] + bta[idx]);
  }
}

// ---------------------------------------------------------------------------
// Row softmax over 2048 keys, fp32 scores -> bf16 probabilities.
// ---------------------------------------------------------------------------
__global__ __launch_bounds__(256) void softmax_kernel(
    const float* __restrict__ S, unsigned short* __restrict__ P) {
  const int tid = threadIdx.x;
  const float* sr = S + (long)blockIdx.x * S_DIM;
  unsigned short* pr = P + (long)blockIdx.x * S_DIM;
  float v[8];
  float mx = -3.4e38f;
#pragma unroll
  for (int i = 0; i < 8; ++i) { v[i] = sr[tid + i * 256]; mx = fmaxf(mx, v[i]); }
#pragma unroll
  for (int off = 16; off; off >>= 1) mx = fmaxf(mx, __shfl_xor(mx, off, 32));
  __shared__ float redM[8], redS[8];
  if ((tid & 31) == 0) redM[tid >> 5] = mx;
  __syncthreads();
  mx = redM[0];
#pragma unroll
  for (int w = 1; w < 8; ++w) mx = fmaxf(mx, redM[w]);
  float sum = 0.f;
#pragma unroll
  for (int i = 0; i < 8; ++i) { v[i] = __expf(v[i] - mx); sum += v[i]; }
#pragma unroll
  for (int off = 16; off; off >>= 1) sum += __shfl_xor(sum, off, 32);
  if ((tid & 31) == 0) redS[tid >> 5] = sum;
  __syncthreads();
  float tot = 0.f;
#pragma unroll
  for (int w = 0; w < 8; ++w) tot += redS[w];
  const float r = 1.0f / tot;
#pragma unroll
  for (int i = 0; i < 8; ++i) pr[tid + i * 256] = f2bf(v[i] * r);
}

// ---------------------------------------------------------------------------
// WMMA bf16 GEMM:  C[m][n] = sum_k A[m][k] * BT[n][k]   (BT pre-transposed)
// Block = 256 thr = 8 waves arranged 2(M) x 4(N); wave tile 64x32
// (4x2 v_wmma_f32_16x16x32_bf16 subtiles); block tile 128x128; K step 32.
// A-frag (16x32 bf16): lane<16 -> M=lane, K kk+0..7 / kk+16..23 (two b128)
// B-frag (32x16 bf16): lane<16 -> N=lane, K kk+0..7 / kk+16..23 (two b128)
// C-frag: VGPR v -> M = v + 8*(lane>=16), N = lane&15
// MODE 0: fp32 row-major      MODE 1: bf16 row-major (+bias)
// MODE 2: bf16 batch-transposed (V -> Vt[b][n][s])
// MODE 3: fp32 = 2*(acc + bias[n] + resid[m][n])   (final output)
// ---------------------------------------------------------------------------
template <int MODE>
__global__ __launch_bounds__(256) void gemm_bf16_kernel(
    const unsigned short* __restrict__ A, const unsigned short* __restrict__ BT,
    void* __restrict__ Cout, const float* __restrict__ bias,
    const float* __restrict__ resid, int K, int lda, int ldb, int ldc,
    long aStride, long bStride, long cStride, int SB) {
  const int tid = threadIdx.x;
  const int lane = tid & 31;
  const int wid = tid >> 5;
  const int z = blockIdx.z;
  A += (long)z * aStride;
  BT += (long)z * bStride;

  const int m0 = blockIdx.x * 128 + (wid & 1) * 64;   // wave: 64 rows
  const int n0 = blockIdx.y * 128 + (wid >> 1) * 32;  // wave: 32 cols

  const int khA = (lane < 16) ? 0 : 8;    // A half-K offset per lane group
  const int khB = (lane < 16) ? 0 : 8;    // B half-K offset per lane group

  const unsigned short* ap[4];
#pragma unroll
  for (int i = 0; i < 4; ++i)
    ap[i] = A + (long)(m0 + i * 16 + (lane & 15)) * lda + khA;
  const unsigned short* bp[2];
#pragma unroll
  for (int j = 0; j < 2; ++j)
    bp[j] = BT + (long)(n0 + j * 16 + (lane & 15)) * ldb + khB;

  v8f acc[4][2];
#pragma unroll
  for (int i = 0; i < 4; ++i)
#pragma unroll
    for (int j = 0; j < 2; ++j) acc[i][j] = {};

  for (int kk = 0; kk < K; kk += 32) {
    FragBF a[4], b[2];
#pragma unroll
    for (int i = 0; i < 4; ++i) {
      a[i].u[0] = *(const uint4*)(ap[i] + kk);
      a[i].u[1] = *(const uint4*)(ap[i] + kk + 16);
    }
#pragma unroll
    for (int j = 0; j < 2; ++j) {
      b[j].u[0] = *(const uint4*)(bp[j] + kk);
      b[j].u[1] = *(const uint4*)(bp[j] + kk + 16);
    }
    __builtin_prefetch((const void*)(ap[0] + kk + 32), 0, 1);  // global_prefetch_b8
    __builtin_prefetch((const void*)(ap[2] + kk + 32), 0, 1);
    __builtin_prefetch((const void*)(bp[0] + kk + 32), 0, 1);
#pragma unroll
    for (int i = 0; i < 4; ++i)
#pragma unroll
      for (int j = 0; j < 2; ++j)
        acc[i][j] = __builtin_amdgcn_wmma_f32_16x16x32_bf16(
            false, a[i].v, false, b[j].v, (short)0, acc[i][j], false, false);
  }

  const int nIdx = lane & 15;
  const int rBase = (lane >> 4) * 8;
#pragma unroll
  for (int tm = 0; tm < 4; ++tm) {
#pragma unroll
    for (int tn = 0; tn < 2; ++tn) {
      const int mT = m0 + tm * 16 + rBase;       // first of 8 consecutive rows
      const int nT = n0 + tn * 16 + nIdx;
      if (MODE == 0) {
        float* C = (float*)Cout + (long)z * cStride;
#pragma unroll
        for (int v = 0; v < 8; ++v) C[(long)(mT + v) * ldc + nT] = acc[tm][tn][v];
      } else if (MODE == 1) {
        unsigned short* C = (unsigned short*)Cout + (long)z * cStride;
        const float bv = bias ? bias[nT] : 0.0f;
#pragma unroll
        for (int v = 0; v < 8; ++v)
          C[(long)(mT + v) * ldc + nT] = f2bf(acc[tm][tn][v] + bv);
      } else if (MODE == 2) {
        // Vt[b][n][s]: 8 consecutive s per lane -> one 16-byte store.
        unsigned short* C = (unsigned short*)Cout;
        const int b = mT / SB;
        const int s = mT - b * SB;
        const float bv = bias ? bias[nT] : 0.0f;
        PackBF8 pk;
#pragma unroll
        for (int v = 0; v < 8; ++v) pk.s[v] = f2bf(acc[tm][tn][v] + bv);
        *(uint4*)&C[((long)b * ldc + nT) * (long)SB + s] = pk.q;
      } else {  // MODE == 3 : final FC + bias + residual, doubled
        float* C = (float*)Cout;
        const float bv = bias[nT];
#pragma unroll
        for (int v = 0; v < 8; ++v) {
          const long idx = (long)(mT + v) * ldc + nT;
          C[idx] = 2.0f * (acc[tm][tn][v] + bv + resid[idx]);
        }
      }
    }
  }
}

// ---------------------------------------------------------------------------
// Host-side launcher.
// ---------------------------------------------------------------------------
extern "C" void kernel_launch(void* const* d_in, const int* in_sizes, int n_in,
                              void* d_out, int out_size, void* d_ws, size_t ws_size,
                              hipStream_t stream) {
  (void)in_sizes; (void)n_in; (void)out_size; (void)ws_size;
  const float* x   = (const float*)d_in[0];
  const float* Wq  = (const float*)d_in[1];
  const float* bq  = (const float*)d_in[2];
  const float* Wk  = (const float*)d_in[3];
  const float* bk  = (const float*)d_in[4];
  const float* Wv  = (const float*)d_in[5];
  const float* bv  = (const float*)d_in[6];
  const float* Wfc = (const float*)d_in[7];
  const float* bfc = (const float*)d_in[8];
  const float* g1  = (const float*)d_in[9];
  const float* b1  = (const float*)d_in[10];
  const float* g2  = (const float*)d_in[11];
  const float* b2  = (const float*)d_in[12];
  float* out = (float*)d_out;

  // Workspace carve-up (bytes).
  char* ws = (char*)d_ws;
  const long WT_BYTES = (long)E_DIM * E_DIM * 2;          //  2 MB each
  const long HS_BYTES = (long)M_TOT * E_DIM * 2;          // 32 MB each
  unsigned short* WqT  = (unsigned short*)(ws);                      // 2 MB
  unsigned short* WkT  = (unsigned short*)(ws + WT_BYTES);           // 2 MB
  unsigned short* WvT  = (unsigned short*)(ws + 2 * WT_BYTES);       // 2 MB
  unsigned short* WfcT = (unsigned short*)(ws + 3 * WT_BYTES);       // 2 MB
  char* p = ws + 4 * WT_BYTES;
  unsigned short* hbuf = (unsigned short*)p; p += HS_BYTES;          // 32 MB (reused for h2)
  unsigned short* Qb   = (unsigned short*)p; p += HS_BYTES;          // 32 MB
  unsigned short* Kb   = (unsigned short*)p; p += HS_BYTES;          // 32 MB
  unsigned short* Vt   = (unsigned short*)p; p += HS_BYTES;          // 32 MB  [b][i][s]
  float*          Sb   = (float*)p;          p += (long)B_DIM * S_DIM * S_DIM * 4;  // 128 MB
  unsigned short* Pb   = (unsigned short*)p; p += (long)B_DIM * S_DIM * S_DIM * 2;  //  64 MB
  unsigned short* Ob   = (unsigned short*)p;                         // 32 MB

  const dim3 blk(256);

  // 1) Weights -> bf16, transposed (BT layout [n][k]).
  wt_convert_kernel<<<dim3(32, 32), blk, 0, stream>>>(Wq,  WqT,  E_DIM, E_DIM);
  wt_convert_kernel<<<dim3(32, 32), blk, 0, stream>>>(Wk,  WkT,  E_DIM, E_DIM);
  wt_convert_kernel<<<dim3(32, 32), blk, 0, stream>>>(Wv,  WvT,  E_DIM, E_DIM);
  wt_convert_kernel<<<dim3(32, 32), blk, 0, stream>>>(Wfc, WfcT, E_DIM, E_DIM);

  // 2) LayerNorm1: x -> h (bf16).
  layernorm_kernel<0><<<dim3(M_TOT), blk, 0, stream>>>(x, g1, b1, hbuf);

  // 3) QKV projections (M=16384, N=1024, K=1024). Block tile 128x128.
  gemm_bf16_kernel<1><<<dim3(128, 8, 1), blk, 0, stream>>>(
      hbuf, WqT, Qb, bq, nullptr, E_DIM, E_DIM, E_DIM, E_DIM, 0, 0, 0, 0);
  gemm_bf16_kernel<1><<<dim3(128, 8, 1), blk, 0, stream>>>(
      hbuf, WkT, Kb, bk, nullptr, E_DIM, E_DIM, E_DIM, E_DIM, 0, 0, 0, 0);
  gemm_bf16_kernel<2><<<dim3(128, 8, 1), blk, 0, stream>>>(
      hbuf, WvT, Vt, bv, nullptr, E_DIM, E_DIM, E_DIM, E_DIM, 0, 0, 0, S_DIM);

  // 4) Scores = Q @ K^T per batch (M=N=2048, K=1024), fp32 out.
  gemm_bf16_kernel<0><<<dim3(16, 16, B_DIM), blk, 0, stream>>>(
      Qb, Kb, Sb, nullptr, nullptr, E_DIM, E_DIM, E_DIM, S_DIM,
      (long)S_DIM * E_DIM, (long)S_DIM * E_DIM, (long)S_DIM * S_DIM, 0);

  // 5) Row softmax -> bf16 P.
  softmax_kernel<<<dim3(B_DIM * S_DIM), blk, 0, stream>>>(Sb, Pb);

  // 6) O = P @ V per batch (M=2048, N=1024, K=2048), bf16 out.
  gemm_bf16_kernel<1><<<dim3(16, 8, B_DIM), blk, 0, stream>>>(
      Pb, Vt, Ob, nullptr, nullptr, S_DIM, S_DIM, S_DIM, E_DIM,
      (long)S_DIM * S_DIM, (long)E_DIM * S_DIM, (long)S_DIM * E_DIM, 0);

  // 7) LayerNorm2: O -> h2 (reuse hbuf).
  layernorm_kernel<1><<<dim3(M_TOT), blk, 0, stream>>>(Ob, g2, b2, hbuf);

  // 8) FC + bias + residual, doubled, fp32 -> d_out.
  gemm_bf16_kernel<3><<<dim3(128, 8, 1), blk, 0, stream>>>(
      hbuf, WfcT, out, bfc, x, E_DIM, E_DIM, E_DIM, E_DIM, 0, 0, 0, 0);
}
